// BasicMultiHeadAttention_78460462563625
// MI455X (gfx1250) — matrix operand order repaired and checked by reading, not compiled, not run on hardware
//
#include <hip/hip_runtime.h>
#include <hip/hip_bf16.h>
#include <stdint.h>

#define EMB 1024
#define HEADS 16
#define HEAD_DIM 64
#define BATCH 4
#define SEQ 2048
#define MTOT (BATCH * SEQ) /* 8192 */

typedef __attribute__((ext_vector_type(16))) __bf16 v16bf;
typedef __attribute__((ext_vector_type(8)))  float  v8f;

__device__ __forceinline__ __bf16 to_bf16(float x)  { return (__bf16)x; }
__device__ __forceinline__ __bf16 to_bf16(__bf16 x) { return x; }

// v_permlane16_b32 based xor-butterfly max over each 16-lane row (VALU, no LDS)
__device__ __forceinline__ float pl16(float v, unsigned s0, unsigned s1) {
  unsigned u = __builtin_bit_cast(unsigned, v);
  unsigned r = __builtin_amdgcn_permlane16(u, u, s0, s1, false, false);
  return __builtin_bit_cast(float, r);
}
__device__ __forceinline__ float rowmax16(float v) {
  v = fmaxf(v, pl16(v, 0x67452301u, 0xEFCDAB89u));  // xor 1
  v = fmaxf(v, pl16(v, 0x54761032u, 0xDCFE98BAu));  // xor 2
  v = fmaxf(v, pl16(v, 0x32107654u, 0xBA98FEDCu));  // xor 4
  v = fmaxf(v, pl16(v, 0xFEDCBA98u, 0x76543210u));  // xor 8
  return v;
}

// ---------------------------------------------------------------------------
// C[M,N] = X[M,K] @ W[N,K]^T   (bf16 WMMA, f32 accumulate)
// Block: 256 threads = 8 waves; block tile 128x128; wave tile 64x32 (4x2 WMMA)
// Register double-buffer: next K-slice global loads overlap current WMMAs.
// ---------------------------------------------------------------------------
template <typename XT, typename OT>
__global__ __launch_bounds__(256) void gemm_xwT(const XT* __restrict__ X,
                                                const float* __restrict__ W,
                                                OT* __restrict__ C,
                                                int M, int N, int K) {
  __shared__ __bf16 lA[128][40];  // 32 K + 8 pad
  __shared__ __bf16 lB[128][40];

  const int tid  = threadIdx.x;
  const int lane = tid & 31;
  const int wid  = tid >> 5;
  const int m0 = blockIdx.y * 128;
  const int n0 = blockIdx.x * 128;
  const int wM = (wid >> 2) * 64;
  const int wN = (wid & 3) * 32;

  const int ldr = tid >> 1;        // 0..127
  const int ldc = (tid & 1) * 16;  // 0 or 16

  const int lm  = lane & 15;
  const int hi  = lane >> 4;
  const int off = hi * 8;          // A-layout K sub-offset (ISA 7.12.2)

  const XT*    xrow = X + (size_t)(m0 + ldr) * K + ldc;
  const float* wrow = W + (size_t)(n0 + ldr) * K + ldc;

  __bf16 xr[16], wr[16];
#pragma unroll
  for (int i = 0; i < 16; ++i) { xr[i] = to_bf16(xrow[i]); wr[i] = to_bf16(wrow[i]); }

  v8f acc[4][2] = {};

  for (int k0 = 0; k0 < K; k0 += 32) {
    // commit current slice to LDS
#pragma unroll
    for (int i = 0; i < 16; ++i) { lA[ldr][ldc + i] = xr[i]; lB[ldr][ldc + i] = wr[i]; }
    __syncthreads();

    // prefetch next slice into registers (overlaps the WMMAs below)
    if (k0 + 32 < K) {
#pragma unroll
      for (int i = 0; i < 16; ++i) {
        xr[i] = to_bf16(xrow[k0 + 32 + i]);
        wr[i] = to_bf16(wrow[k0 + 32 + i]);
      }
    }

    v16bf bfrag[2];
#pragma unroll
    for (int nt = 0; nt < 2; ++nt) {
      const __bf16* bp = &lB[wN + nt * 16 + lm][0];
#pragma unroll
      for (int i = 0; i < 8; ++i) { bfrag[nt][i] = bp[off + i]; bfrag[nt][8 + i] = bp[16 + off + i]; }
    }
#pragma unroll
    for (int mt = 0; mt < 4; ++mt) {
      const __bf16* ap = &lA[wM + mt * 16 + lm][0];
      v16bf a;
#pragma unroll
      for (int i = 0; i < 8; ++i) { a[i] = ap[off + i]; a[8 + i] = ap[16 + off + i]; }
#pragma unroll
      for (int nt = 0; nt < 2; ++nt)
        acc[mt][nt] = __builtin_amdgcn_wmma_f32_16x16x32_bf16(
            false, a, false, bfrag[nt], (short)0, acc[mt][nt], false, false);
    }
    __syncthreads();
  }

  const int rhalf = hi * 8;
#pragma unroll
  for (int mt = 0; mt < 4; ++mt)
#pragma unroll
    for (int nt = 0; nt < 2; ++nt)
#pragma unroll
      for (int r = 0; r < 8; ++r) {
        const int row = m0 + wM + mt * 16 + r + rhalf;
        const int col = n0 + wN + nt * 16 + lm;
        C[(size_t)row * N + col] = (OT)acc[mt][nt][r];
      }
}

// ---------------------------------------------------------------------------
// Vt[(b*H+h)][d][s] = V[(b*S+s)][h*64+d]   (bf16, 64x64 LDS tiles)
// ---------------------------------------------------------------------------
__global__ __launch_bounds__(256) void transpose_v(const __bf16* __restrict__ V,
                                                   __bf16* __restrict__ Vt) {
  __shared__ __bf16 sh[64][72];
  const int tid = threadIdx.x;
  const int st  = blockIdx.x & 31;
  const int bh  = blockIdx.x >> 5;
  const int b = bh >> 4, h = bh & 15;
  const int kt = st * 64;
  {
    const int r = tid >> 2, c0 = (tid & 3) * 16;
    const __bf16* src = V + (size_t)(b * SEQ + kt + r) * EMB + h * HEAD_DIM + c0;
#pragma unroll
    for (int i = 0; i < 16; ++i) sh[r][c0 + i] = src[i];
  }
  __syncthreads();
  {
    const int d = tid >> 2, k0 = (tid & 3) * 16;
    __bf16* dst = Vt + ((size_t)bh * HEAD_DIM + d) * SEQ + kt + k0;
#pragma unroll
    for (int i = 0; i < 16; ++i) dst[i] = sh[k0 + i][d];
  }
}

// ---------------------------------------------------------------------------
// Flash attention. Wave owns 16 query rows; processes keys in 128-wide blocks:
//   16 WMMA for QK^T, one permlane16 max-reduce per row per block,
//   row-sums via WMMA against a ones matrix, 16 WMMA for PV.
// ---------------------------------------------------------------------------
__global__ __launch_bounds__(256) void attn_fwd(const __bf16* __restrict__ Q,
                                                const __bf16* __restrict__ Kb,
                                                const __bf16* __restrict__ Vt,
                                                __bf16* __restrict__ Ctx) {
  __shared__ __bf16 Pst[8][16][136];  // per-wave P tile (16 rows x 128 keys)

  const int tid = threadIdx.x, lane = tid & 31, wid = tid >> 5;
  const int qb = blockIdx.x & 15;
  const int bh = blockIdx.x >> 4;
  const int b = bh >> 4, h = bh & 15;
  const int q0 = qb * 128 + wid * 16;

  const int lm  = lane & 15;
  const int hi  = lane >> 4;
  const int off = hi * 8;

  // Q fragments (A layout)
  const __bf16* qp = Q + (size_t)(b * SEQ + q0 + lm) * EMB + h * HEAD_DIM;
  v16bf aq0, aq1;
#pragma unroll
  for (int i = 0; i < 8; ++i) {
    aq0[i]     = qp[off + i];
    aq0[8 + i] = qp[16 + off + i];
    aq1[i]     = qp[32 + off + i];
    aq1[8 + i] = qp[48 + off + i];
  }

  const __bf16* Kbase = Kb + (size_t)(b * SEQ) * EMB + h * HEAD_DIM + hi * 8;
  const __bf16* Vbase = Vt + (size_t)bh * HEAD_DIM * SEQ;

  v16bf bones;  // ones matrix for row sums via WMMA
#pragma unroll
  for (int i = 0; i < 16; ++i) bones[i] = (__bf16)1.0f;

  v8f acc[4] = {};
  v8f lacc = {};     // softmax denominators per row (replicated across cols)
  float mrow[8];
#pragma unroll
  for (int r = 0; r < 8; ++r) mrow[r] = -3.0e38f;
  const float scale = 0.125f;  // 1/sqrt(64)

  for (int kk = 0; kk < SEQ; kk += 128) {
    // ---- scores: 8 subtiles of 16 keys, K-depth 64 (2 chained WMMA each)
    v8f sc[8];
#pragma unroll
    for (int j = 0; j < 8; ++j) {
      const __bf16* kp = Kbase + (size_t)(kk + j * 16 + lm) * EMB;
      __builtin_prefetch(kp + 128 * EMB, 0, 1);  // next key block -> L2/L0
      v16bf bk0, bk1;
#pragma unroll
      for (int i = 0; i < 8; ++i) { bk0[i] = kp[i]; bk0[8 + i] = kp[16 + i]; bk1[i] = kp[32 + i]; bk1[8 + i] = kp[48 + i]; }
      v8f z = {};
      z = __builtin_amdgcn_wmma_f32_16x16x32_bf16(false, aq0, false, bk0, (short)0, z, false, false);
      z = __builtin_amdgcn_wmma_f32_16x16x32_bf16(false, aq1, false, bk1, (short)0, z, false, false);
      sc[j] = z;
    }

    // ---- one online-softmax max update per row per 128 keys
    float mnew[8];
#pragma unroll
    for (int r = 0; r < 8; ++r) {
      float v = sc[0][r];
#pragma unroll
      for (int j = 1; j < 8; ++j) v = fmaxf(v, sc[j][r]);
      v = rowmax16(v * scale);
      mnew[r] = fmaxf(mrow[r], v);
      const float alpha = __expf(mrow[r] - mnew[r]);
      mrow[r] = mnew[r];
      lacc[r] *= alpha;
#pragma unroll
      for (int nt = 0; nt < 4; ++nt) acc[nt][r] *= alpha;
    }

    // ---- P = exp(s - m), staged to LDS as bf16 in row-major 16x128
#pragma unroll
    for (int j = 0; j < 8; ++j)
#pragma unroll
      for (int r = 0; r < 8; ++r) {
        const float p = __expf(sc[j][r] * scale - mnew[r]);
        Pst[wid][r + hi * 8][j * 16 + lm] = (__bf16)p;
      }

    // ---- P tiles (16x32) : row-sum WMMA (ones) + PV WMMAs
#pragma unroll
    for (int t = 0; t < 4; ++t) {
      v16bf ap;
      const __bf16* pp = &Pst[wid][lm][t * 32];
#pragma unroll
      for (int i = 0; i < 8; ++i) { ap[i] = pp[off + i]; ap[8 + i] = pp[16 + off + i]; }

      lacc = __builtin_amdgcn_wmma_f32_16x16x32_bf16(
          false, ap, false, bones, (short)0, lacc, false, false);

#pragma unroll
      for (int nt = 0; nt < 4; ++nt) {
        const __bf16* vp = Vbase + (size_t)(nt * 16 + lm) * SEQ + kk + t * 32 + hi * 8;
        v16bf bv;
#pragma unroll
        for (int i = 0; i < 8; ++i) { bv[i] = vp[i]; bv[8 + i] = vp[16 + i]; }
        acc[nt] = __builtin_amdgcn_wmma_f32_16x16x32_bf16(
            false, ap, false, bv, (short)0, acc[nt], false, false);
      }
    }
  }

  // ---- normalize and write ctx[b,s,h,d]
#pragma unroll
  for (int nt = 0; nt < 4; ++nt)
#pragma unroll
    for (int r = 0; r < 8; ++r) {
      const int row = q0 + r + hi * 8;
      const float v = acc[nt][r] / lacc[r];
      Ctx[(size_t)(b * SEQ + row) * EMB + h * HEAD_DIM + nt * 16 + lm] = (__bf16)v;
    }
}

// ---------------------------------------------------------------------------
extern "C" void kernel_launch(void* const* d_in, const int* in_sizes, int n_in,
                              void* d_out, int out_size, void* d_ws, size_t ws_size,
                              hipStream_t stream) {
  const float* query = (const float*)d_in[0];
  const float* key   = (const float*)d_in[1];
  const float* value = (const float*)d_in[2];
  const float* Wq    = (const float*)d_in[3];
  const float* Wk    = (const float*)d_in[4];
  const float* Wv    = (const float*)d_in[5];
  const float* Wo    = (const float*)d_in[6];
  float* out = (float*)d_out;

  const size_t NEL = (size_t)MTOT * EMB;  // 16 MB per bf16 buffer
  __bf16* Qb  = (__bf16*)d_ws;
  __bf16* Kb  = Qb + NEL;
  __bf16* Vb  = Kb + NEL;
  __bf16* Vt  = Vb + NEL;
  __bf16* Ctx = Vt + NEL;

  dim3 blk(256);
  dim3 gg(EMB / 128, MTOT / 128);  // 8 x 64

  gemm_xwT<float, __bf16><<<gg, blk, 0, stream>>>(query, Wq, Qb, MTOT, EMB, EMB);
  gemm_xwT<float, __bf16><<<gg, blk, 0, stream>>>(key,   Wk, Kb, MTOT, EMB, EMB);
  gemm_xwT<float, __bf16><<<gg, blk, 0, stream>>>(value, Wv, Vb, MTOT, EMB, EMB);

  transpose_v<<<BATCH * HEADS * (SEQ / 64), blk, 0, stream>>>(Vb, Vt);

  attn_fwd<<<BATCH * HEADS * (SEQ / 128), blk, 0, stream>>>(Qb, Kb, Vt, Ctx);

  gemm_xwT<__bf16, float><<<gg, blk, 0, stream>>>(Ctx, Wo, out, MTOT, EMB, EMB);
}
